// LinearSplineCoupling_70600672412077
// MI455X (gfx1250) — compile-verified, matrix operand order checked by reading, not run on hardware
//
#include <hip/hip_runtime.h>

// ---------------------------------------------------------------------------
// Types
// ---------------------------------------------------------------------------
typedef __attribute__((ext_vector_type(16))) __bf16 bf16x16;
typedef __attribute__((ext_vector_type(8)))  __bf16 bf16x8;
typedef __attribute__((ext_vector_type(8)))  float  f32x8;

__device__ __forceinline__ unsigned short f2bf(float f) {
    unsigned int u = __builtin_bit_cast(unsigned int, f);
    u += 0x7FFFu + ((u >> 16) & 1u);            // round-to-nearest-even
    return (unsigned short)(u >> 16);
}

#define SHUF16(lo, hi) __builtin_shufflevector((lo), (hi), 0,1,2,3,4,5,6,7,8,9,10,11,12,13,14,15)

// Problem constants
#define BB   32
#define HH   64
#define WW   64
#define HWW  4096
#define CIN  32      // identity half channels (conv1 input)
#define HID  128     // hidden channels
#define COUT 96      // 32 * 3 bins

// Fragment counts (512 halfs per 16x32 A fragment)
#define N1FRAG (8*9*512)        // conv1: 8 ch-tiles x 9 taps
#define N2FRAG (6*9*4*512)      // conv2: 6 ch-tiles x 9 taps x 4 K-blocks

// ---------------------------------------------------------------------------
// Weight repack into WMMA A-fragment layout (bf16, lane-major, 32B per lane)
// lane L<16: row M=L, j<8 -> K=j,       j>=8 -> K=8+j   (i.e. 16..23)
// lane L>=16: row M=L-16, j<8 -> K=8+j, j>=8 -> K=16+j  (i.e. 24..31)
// ---------------------------------------------------------------------------
__global__ __launch_bounds__(256) void repack_weights(
        const float* __restrict__ W1, const float* __restrict__ W2,
        unsigned short* __restrict__ w1f, unsigned short* __restrict__ w2f) {
    int tid = blockIdx.x * 256 + threadIdx.x;
    if (tid < N1FRAG) {
        int e = tid;
        int j    = e & 15;
        int lane = (e >> 4) & 31;
        int kk   = (e >> 9) % 9;
        int ct   = e / (512 * 9);
        int M = lane & 15, hi = lane >> 4;
        int kp = (j < 8) ? (hi * 8 + j) : (8 + hi * 8 + j);
        int o  = ct * 16 + M;
        int dy = kk / 3, dx = kk % 3;
        w1f[e] = f2bf(W1[((o * CIN + kp) * 3 + dy) * 3 + dx]);
    } else if (tid < N1FRAG + N2FRAG) {
        int e = tid - N1FRAG;
        int j    = e & 15;
        int lane = (e >> 4) & 31;
        int cb   = (e >> 9) & 3;
        int kk   = (e >> 11) % 9;
        int ct   = e / (2048 * 9);
        int M = lane & 15, hi = lane >> 4;
        int kp  = (j < 8) ? (hi * 8 + j) : (8 + hi * 8 + j);
        int cin = cb * 32 + kp;
        int o   = ct * 16 + M;
        int dy = kk / 3, dx = kk % 3;
        w2f[e] = f2bf(W2[((o * HID + cin) * 3 + dy) * 3 + dx]);
    }
}

// ---------------------------------------------------------------------------
// conv1: 32 -> 128 channels, 3x3 SAME, +bias, ReLU.
// Output hbuf is NHWC bf16 [B,H,W,128] so each lane's 8 accum rows pack into
// one contiguous global_store_b128, and conv2 staging is fully contiguous.
// WG = 256 thr (8 waves). Each WG: 16 pixels of one row; wave w = ch-tile w.
// ---------------------------------------------------------------------------
__global__ __launch_bounds__(256) void conv1_relu(
        const float* __restrict__ x, const float* __restrict__ b1,
        const unsigned short* __restrict__ w1f,
        unsigned short* __restrict__ hbuf) {
    __shared__ __align__(16) unsigned short smem[3 * 18 * 32];   // [row][col][ch]
    int wg = blockIdx.x;
    int pt = wg & 3;                 // 16-pixel tile in row
    int hr = (wg >> 2) & 63;
    int b  = wg >> 8;
    int tid = threadIdx.x;

    for (int i = tid; i < 3 * 18 * 32; i += 256) {
        int ch  = i & 31;
        int col = (i >> 5) % 18;
        int row = i / (18 * 32);
        int gy = hr - 1 + row;
        int gx = pt * 16 - 1 + col;
        float v = 0.f;
        if ((unsigned)gy < 64u && (unsigned)gx < 64u)
            v = x[((b * 64 + ch) * HH + gy) * WW + gx];   // identity half = ch 0..31
        smem[i] = f2bf(v);
    }
    __syncthreads();

    int wave = tid >> 5;
    int lane = tid & 31;
    int Ncol = lane & 15;
    int hi   = lane >> 4;
    f32x8 acc = {};
#pragma unroll
    for (int kk = 0; kk < 9; ++kk) {
        int dy = kk / 3, dx = kk % 3;
        bf16x16 a = *(const bf16x16*)(w1f + (wave * 9 + kk) * 512 + lane * 16);
        int lb = (dy * 18 + Ncol + dx) * 32 + hi * 8;
        bf16x8 blo = *(const bf16x8*)(smem + lb);
        bf16x8 bhi = *(const bf16x8*)(smem + lb + 16);
        bf16x16 bfr = SHUF16(blo, bhi);
        acc = __builtin_amdgcn_wmma_f32_16x16x32_bf16(
                false, a, false, bfr, (short)0, acc, false, false);
    }
    // bias + relu + pack 8 consecutive channels -> one b128 store (NHWC)
    int chbase = wave * 16 + hi * 8;
    unsigned int pk[4];
#pragma unroll
    for (int r = 0; r < 4; ++r) {
        float v0 = fmaxf(acc[2 * r]     + b1[chbase + 2 * r],     0.f);
        float v1 = fmaxf(acc[2 * r + 1] + b1[chbase + 2 * r + 1], 0.f);
        pk[r] = (unsigned)f2bf(v0) | ((unsigned)f2bf(v1) << 16);
    }
    size_t o = ((size_t)((b * HH + hr) * WW + pt * 16 + Ncol)) * HID + chbase;
    *(uint4*)(hbuf + o) = make_uint4(pk[0], pk[1], pk[2], pk[3]);
}

// ---------------------------------------------------------------------------
// conv2: 128 -> 96 channels, 3x3 SAME, +bias, f32 output (spline params).
// Input patch staged global->LDS with ASYNC-to-LDS b128 copies (no VGPR
// round-trip); halo chunks zero-filled with ds_store_b128.
// WG = 384 thr (12 waves) covering 32 pixels: wave = (ch-tile 0..5, pix-sub 0..1)
// ---------------------------------------------------------------------------
__global__ __launch_bounds__(384) void conv2_k(
        const unsigned short* __restrict__ hbuf, const float* __restrict__ b2,
        const unsigned short* __restrict__ w2f, float* __restrict__ params) {
    __shared__ __align__(16) unsigned short smem[3 * 34 * 128];  // [row][col][ch]
    int wg = blockIdx.x;
    int pt = wg & 1;                 // 32-pixel tile in row
    int hr = (wg >> 1) & 63;
    int b  = wg >> 7;
    int tid = threadIdx.x;

    // stage 3 x 34 x 128 bf16 patch: 16B chunks, async global->LDS
    unsigned lds_base = (unsigned)(unsigned long long)(uintptr_t)(void*)smem;
    const int NCHUNK = 3 * 34 * 16;                       // 1632 chunks of 16B
    for (int i = tid; i < NCHUNK; i += 384) {
        int c   = i & 15;            // chunk within 128 channels (8 ch each)
        int col = (i >> 4) % 34;
        int row = i / (34 * 16);
        int gy = hr - 1 + row;
        int gx = pt * 32 - 1 + col;
        int lelem = (row * 34 + col) * 128 + c * 8;       // ushort index in smem
        if ((unsigned)gy < 64u && (unsigned)gx < 64u) {
            const unsigned short* gp =
                hbuf + ((size_t)((b * HH + gy) * WW + gx)) * HID + c * 8;
            unsigned loff = lds_base + (unsigned)(lelem * 2);
            unsigned long long ga = (unsigned long long)(uintptr_t)gp;
            asm volatile("global_load_async_to_lds_b128 %0, %1, off"
                         :: "v"(loff), "v"(ga) : "memory");
        } else {
            *(uint4*)(smem + lelem) = make_uint4(0u, 0u, 0u, 0u);
        }
    }
    asm volatile("s_wait_asynccnt 0" ::: "memory");
    __syncthreads();

    int wave = tid >> 5;
    int lane = tid & 31;
    int ct   = wave % 6;             // out-channel tile
    int ps   = wave / 6;             // pixel sub-tile (0/1)
    int Ncol = lane & 15;
    int hi   = lane >> 4;
    int colb = ps * 16;
    f32x8 acc = {};
#pragma unroll
    for (int kk = 0; kk < 9; ++kk) {
        int dy = kk / 3, dx = kk % 3;
        int lrow = (dy * 34 + colb + Ncol + dx) * 128;
#pragma unroll
        for (int cb = 0; cb < 4; ++cb) {
            bf16x16 a = *(const bf16x16*)(w2f + ((ct * 9 + kk) * 4 + cb) * 512 + lane * 16);
            int lb = lrow + cb * 32 + hi * 8;
            bf16x8 blo = *(const bf16x8*)(smem + lb);
            bf16x8 bhi = *(const bf16x8*)(smem + lb + 16);
            bf16x16 bfr = SHUF16(blo, bhi);
            acc = __builtin_amdgcn_wmma_f32_16x16x32_bf16(
                    false, a, false, bfr, (short)0, acc, false, false);
        }
    }
#pragma unroll
    for (int r = 0; r < 8; ++r) {
        int M  = r + hi * 8;
        int ch = ct * 16 + M;
        params[((b * COUT + ch) * HH + hr) * WW + pt * 32 + colb + Ncol] = acc[r] + b2[ch];
    }
}

// ---------------------------------------------------------------------------
// Spline transform + per-batch log|det| reduction
// ---------------------------------------------------------------------------
__global__ __launch_bounds__(256) void spline_k(
        const float* __restrict__ x, const float* __restrict__ params,
        float* __restrict__ out, float* __restrict__ obj) {
    int b     = blockIdx.x >> 9;                 // 512 blocks per batch
    int local = (blockIdx.x & 511) * 256 + threadIdx.x;   // 0 .. 131071
    int c   = local >> 12;                       // 0..31
    int pix = local & 4095;

    float xin = x[(b * 64 + 32 + c) * HWW + pix];    // transform half = ch 32..63
    const float* pb = params + (size_t)(b * COUT + c * 3) * HWW + pix;
    float p0 = pb[0], p1 = pb[HWW], p2 = pb[2 * HWW];

    float m  = fmaxf(p0, fmaxf(p1, p2));
    float e0 = __expf(p0 - m), e1 = __expf(p1 - m), e2 = __expf(p2 - m);
    float inv = 1.f / (e0 + e1 + e2);
    float q0 = e0 * inv, q1 = e1 * inv, q2 = e2 * inv;

    bool inside = (xin >= -1.f) && (xin <= 1.f);
    float x01 = (xin + 1.f) * 0.5f;
    float pos = fminf(fmaxf(x01, 0.f), 1.f) * 3.f;
    int bin = (int)floorf(pos);
    bin = bin < 0 ? 0 : (bin > 2 ? 2 : bin);
    float alpha = pos - (float)bin;
    float pdfb = (bin == 0) ? q0 : ((bin == 1) ? q1 : q2);
    float cdfb = (bin == 0) ? 0.f : ((bin == 1) ? q0 : q0 + q1);
    float o01 = fminf(fmaxf(cdfb + alpha * pdfb, 0.f), 1.f);
    float oval = o01 * 2.f - 1.f;
    float lad  = __logf(pdfb) + 1.0986122886681098f;   // + log(3)
    oval = inside ? oval : xin;
    lad  = inside ? lad : 0.f;

    out[(b * 64 + 32 + c) * HWW + pix] = oval;

    // block reduction of lad
    for (int off = 16; off; off >>= 1) lad += __shfl_down(lad, off);
    __shared__ float red[8];
    if ((threadIdx.x & 31) == 0) red[threadIdx.x >> 5] = lad;
    __syncthreads();
    if (threadIdx.x < 8) {
        float v = red[threadIdx.x];
        for (int off = 4; off; off >>= 1) v += __shfl_down(v, off);
        if (threadIdx.x == 0) atomicAdd(&obj[b], v);
    }
}

__global__ void init_obj(const float* __restrict__ obj_in, float* __restrict__ obj) {
    int i = threadIdx.x;
    if (i < BB) obj[i] = obj_in[i];
}

__global__ __launch_bounds__(256) void copy_identity(
        const float4* __restrict__ x4, float4* __restrict__ o4) {
    int g = blockIdx.x * 256 + threadIdx.x;
    if (g >= BB * 32768) return;           // 32 ch * 4096 px / 4 per batch
    int b = g >> 15, r = g & 32767;
    o4[b * 65536 + r] = x4[b * 65536 + r];
}

// ---------------------------------------------------------------------------
// Launch
// ---------------------------------------------------------------------------
extern "C" void kernel_launch(void* const* d_in, const int* in_sizes, int n_in,
                              void* d_out, int out_size, void* d_ws, size_t ws_size,
                              hipStream_t stream) {
    const float* x   = (const float*)d_in[0];
    const float* obj = (const float*)d_in[1];
    const float* W1  = (const float*)d_in[2];
    const float* b1  = (const float*)d_in[3];
    const float* W2  = (const float*)d_in[4];
    const float* b2  = (const float*)d_in[5];

    // workspace layout
    char* ws = (char*)d_ws;
    unsigned short* w1f   = (unsigned short*)(ws);                         //   73,728 B
    unsigned short* w2f   = (unsigned short*)(ws + 73728);                 //  221,184 B
    unsigned short* hbuf  = (unsigned short*)(ws + 294912);                // 33.55 MB bf16 NHWC [B,H,W,128]
    float*          param = (float*)(ws + 294912 + 33554432);              // 50.33 MB f32 [B,96,H,W]

    float* out_main = (float*)d_out;
    float* out_obj  = out_main + (size_t)BB * 64 * HWW;    // objective after main tensor

    repack_weights<<<(N1FRAG + N2FRAG + 255) / 256, 256, 0, stream>>>(W1, W2, w1f, w2f);
    conv1_relu   <<<BB * HH * (WW / 16), 256, 0, stream>>>(x, b1, w1f, hbuf);
    conv2_k      <<<BB * HH * (WW / 32), 384, 0, stream>>>(hbuf, b2, w2f, param);
    init_obj     <<<1, 32, 0, stream>>>(obj, out_obj);
    copy_identity<<<(BB * 32768 + 255) / 256, 256, 0, stream>>>((const float4*)x, (float4*)out_main);
    spline_k     <<<BB * 512, 256, 0, stream>>>(x, param, out_main, out_obj);
}
